// EdgeScorer_63763084476987
// MI455X (gfx1250) — matrix operand order repaired and checked by reading, not compiled, not run on hardware
//
#include <hip/hip_runtime.h>
#include <hip/hip_bf16.h>

// ---- types for CDNA5 WMMA ----
typedef __attribute__((ext_vector_type(16))) __bf16 v16bf;
typedef __attribute__((ext_vector_type(8)))  float  v8f;

union FragBF16 {            // 16 bf16 = 32 bytes = 8 VGPRs
    v16bf v;
    uint4 q[2];
};

__device__ __forceinline__ unsigned short f32_to_bf16_rne(float f) {
    union { float f; unsigned u; } v; v.f = f;
    unsigned u = v.u;
    unsigned r = u + 0x7FFFu + ((u >> 16) & 1u);   // round-to-nearest-even
    return (unsigned short)(r >> 16);
}

#define NODE_DIM 256
#define IN_DIM   512
#define HIDDEN   512

// ---- CDNA5 async global->LDS (ASYNCcnt-tracked, no VGPR round trip) ----
__device__ __forceinline__ void async_load_b128(unsigned ldsByteAddr, const void* gptr) {
    asm volatile("global_load_async_to_lds_b128 %0, %1, off"
                 :: "v"(ldsByteAddr), "v"(gptr) : "memory");
}
__device__ __forceinline__ void wait_asynccnt0() {
    asm volatile("s_wait_asynccnt 0x0" ::: "memory");
}

// ---- prep: x (f32) -> bf16 ----
__global__ void cvt_x_bf16(const float* __restrict__ x,
                           unsigned short* __restrict__ xb, int n) {
    int i = blockIdx.x * blockDim.x + threadIdx.x;
    int stride = gridDim.x * blockDim.x;
    for (; i < n; i += stride) xb[i] = f32_to_bf16_rne(x[i]);
}

// ---- prep: W1 (f32 [512][512]) -> W1^T bf16 ([n][k]) ----
__global__ void cvt_w1t_bf16(const float* __restrict__ w1,
                             unsigned short* __restrict__ w1t) {
    int i = blockIdx.x * blockDim.x + threadIdx.x;   // 0 .. 512*512-1
    int n = i >> 9;          // hidden index
    int k = i & 511;         // input index
    w1t[i] = f32_to_bf16_rne(w1[k * HIDDEN + n]);    // w1t[n][k] = w1[k][n]
}

// ---- main: per-WG 128 edges; 8 waves x 16-edge strips; bf16 WMMA GEMM ----
__global__ void __launch_bounds__(256)
edge_scorer_wmma(const unsigned short* __restrict__ xb,    // [N_NODES][256] bf16
                 const unsigned short* __restrict__ w1t,   // [512][512] bf16, [n][k]
                 const int*            __restrict__ ei,    // [2][E]
                 const float*          __restrict__ b1,    // [512]
                 const float*          __restrict__ w2,    // [512]
                 const float*          __restrict__ b2,    // [1]
                 float*                __restrict__ out,   // [E]
                 int E) {
    // double-buffered 2 x 16KB W1^T tiles
    __shared__ __align__(16) unsigned short ldsB[2][16 * IN_DIM];

    const int tid  = threadIdx.x;
    const int wave = tid >> 5;
    const int lane = tid & 31;
    const int nsel = lane & 15;          // N / M-row selector within half-wave
    const int half = lane >> 4;          // 0: lanes 0-15, 1: lanes 16-31

    const int eBase = blockIdx.x * 128 + wave * 16;

    // ---- gather A fragments (16 edges x K=512) directly into registers ----
    int e  = eBase + nsel;
    int eg = e < E ? e : (E - 1);
    const int r = ei[eg];
    const int c = ei[E + eg];

    const uint4* prow = (const uint4*)(xb + (size_t)r * NODE_DIM);
    const uint4* pcol = (const uint4*)(xb + (size_t)c * NODE_DIM);
    const int sub = half * 8;            // lane-half K offset inside each 32-block

    FragBF16 a[16];
#pragma unroll
    for (int kt = 0; kt < 16; ++kt) {
        const uint4* src = (kt < 8) ? prow : pcol;
        const int kb = (kt & 7) * 32 + sub;          // bf16 offset within node row
        a[kt].q[0] = src[kb >> 3];                   // K = kb .. kb+7
        a[kt].q[1] = src[(kb + 16) >> 3];            // K = kb+16 .. kb+23
    }

    // ---- async prefetch of tile 0 into buffer 0 ----
    {
        const char* src = (const char*)(w1t);
        const unsigned lbase = (unsigned)(unsigned long long)(const void*)&ldsB[0][0];
#pragma unroll
        for (int i = 0; i < 4; ++i) {
            const int off = (tid + i * 256) * 16;
            async_load_b128(lbase + off, src + off);
        }
    }

    float score[8] = {0.f, 0.f, 0.f, 0.f, 0.f, 0.f, 0.f, 0.f};

    // ---- loop over 32 N-tiles of 16 hidden units ----
    for (int nt = 0; nt < 32; ++nt) {
        const int p = nt & 1;
        wait_asynccnt0();                // my wave's async loads for tile nt done
        __syncthreads();                 // all waves' loads done; prev reads done

        if (nt < 31) {                   // async prefetch tile nt+1 into buffer p^1
            const char* src = (const char*)(w1t + (size_t)(nt + 1) * 16 * IN_DIM);
            const unsigned lbase =
                (unsigned)(unsigned long long)(const void*)&ldsB[p ^ 1][0];
#pragma unroll
            for (int i = 0; i < 4; ++i) {
                const int off = (tid + i * 256) * 16;
                async_load_b128(lbase + off, src + off);
            }
        }

        v8f acc0 = {};
        v8f acc1 = {};
        const float b1v = b1[nt * 16 + nsel];
        const float w2v = w2[nt * 16 + nsel];
        const unsigned short* bufp = &ldsB[p][0];

#pragma unroll
        for (int kt = 0; kt < 16; kt += 2) {
            FragBF16 b0, b1f;
            const int koff0 = kt * 32 + half * 8;
            const int koff1 = (kt + 1) * 32 + half * 8;
            const uint4* p0 = (const uint4*)(bufp + nsel * IN_DIM + koff0);
            const uint4* p1 = (const uint4*)(bufp + nsel * IN_DIM + koff1);
            b0.q[0] = p0[0];  b0.q[1] = p0[2];
            b1f.q[0] = p1[0]; b1f.q[1] = p1[2];
            acc0 = __builtin_amdgcn_wmma_f32_16x16x32_bf16(
                       false, a[kt].v,     false, b0.v,  (short)0, acc0, false, false);
            acc1 = __builtin_amdgcn_wmma_f32_16x16x32_bf16(
                       false, a[kt + 1].v, false, b1f.v, (short)0, acc1, false, false);
        }

        // epilogue: +b1, ReLU, *W2[n], reduce over N (16 lanes per half-wave)
#pragma unroll
        for (int j = 0; j < 8; ++j) {
            float h = (acc0[j] + acc1[j]) + b1v;
            h = h > 0.f ? h : 0.f;
            float pv = h * w2v;
            pv += __shfl_xor(pv, 1, 32);
            pv += __shfl_xor(pv, 2, 32);
            pv += __shfl_xor(pv, 4, 32);
            pv += __shfl_xor(pv, 8, 32);
            score[j] += pv;              // row M = j + 8*half
        }
    }

    const float b2v = b2[0];
    if (nsel == 0) {                     // lanes 0 and 16 write 8 rows each
        const int mBase = half * 8;
#pragma unroll
        for (int j = 0; j < 8; ++j) {
            const int eo = eBase + mBase + j;
            if (eo < E) out[eo] = score[j] + b2v;
        }
    }
}

extern "C" void kernel_launch(void* const* d_in, const int* in_sizes, int n_in,
                              void* d_out, int out_size, void* d_ws, size_t ws_size,
                              hipStream_t stream) {
    const float* x  = (const float*)d_in[0];
    const int*   ei = (const int*)d_in[1];
    const float* W1 = (const float*)d_in[2];
    const float* b1 = (const float*)d_in[3];
    const float* W2 = (const float*)d_in[4];
    const float* b2 = (const float*)d_in[5];
    float* out = (float*)d_out;

    const int nx = in_sizes[0];          // N_NODES * 256
    const int E  = in_sizes[1] / 2;      // edge_index is [2][E]

    unsigned short* xb  = (unsigned short*)d_ws;           // bf16 node features
    unsigned short* w1t = xb + nx;                         // bf16 W1^T [512][512]

    cvt_x_bf16<<<4096, 256, 0, stream>>>(x, xb, nx);
    cvt_w1t_bf16<<<(IN_DIM * HIDDEN) / 256, 256, 0, stream>>>(W1, w1t);

    const int nWG = (E + 127) / 128;
    edge_scorer_wmma<<<nWG, 256, 0, stream>>>(xb, w1t, ei, b1, W2, b2, out, E);
}
